// CausalSelfAttention_35536559407317
// MI455X (gfx1250) — matrix operand order repaired and checked by reading, not compiled, not run on hardware
//
#include <hip/hip_runtime.h>
#include <hip/hip_bf16.h>

// ---------------------------------------------------------------------------
// Landmark (grouped-softmax) causal self-attention for gfx1250 (MI455X).
// B=1, T=2048, C=768, H=12, hd=64, M=64, MEM_FREQ=64.
//
// f32->f16 convert (weights pre-transposed) -> QKV GEMM (WMMA f16, 128x128
// tiles, double-buffered async-to-LDS pipeline, fused bias + scatter:
// Q (h,t,d)*0.125, K (h,t,d), Vt (h,d,t)) -> gated flash attention with
// landmark grouped softmax (WMMA f16 QK^T and PV, K/V section tiles
// double-buffered via async-to-LDS) -> output projection (WMMA f16, f32 out).
// ---------------------------------------------------------------------------

typedef __attribute__((ext_vector_type(16))) _Float16 v16h;
typedef __attribute__((ext_vector_type(8)))  _Float16 v8h;
typedef __attribute__((ext_vector_type(4)))  _Float16 v4h;
typedef __attribute__((ext_vector_type(8)))  float    v8f;
typedef __attribute__((ext_vector_type(4)))  int      v4i;

#define T_SEQ 2048
#define C_DIM 768
#define H_NUM 12
#define HD    64
#define NSEC  32          // T / MEM_FREQ

#if defined(__has_builtin)
#if __has_builtin(__builtin_amdgcn_global_load_async_to_lds_b128)
#define HAS_ASYNC_LDS 1
#endif
#if __has_builtin(__builtin_amdgcn_s_wait_asynccnt)
#define HAS_WAIT_ASYNC 1
#endif
#endif

// 16-byte global->LDS copy; async (ASYNCcnt-tracked) when available.
__device__ inline void cp_b128(_Float16* ldst, const _Float16* gsrc) {
#if defined(HAS_ASYNC_LDS)
  __builtin_amdgcn_global_load_async_to_lds_b128(
      (__attribute__((address_space(1))) v4i*)gsrc,
      (__attribute__((address_space(3))) v4i*)ldst, 0, 0);
#else
  *(v8h*)ldst = *(const v8h*)gsrc;
#endif
}

// Wait until this wave's outstanding async-to-LDS count <= N.
// Per-wave, in-order completion: N=4 means "everything except the 4 most
// recently issued tiles' transfers is done".
template <int N>
__device__ inline void wait_async() {
#if defined(HAS_ASYNC_LDS)
#if defined(HAS_WAIT_ASYNC)
  __builtin_amdgcn_s_wait_asynccnt(N);
#else
  asm volatile("s_wait_asynccnt %0" :: "i"(N) : "memory");
#endif
#endif
}

__device__ inline v8f wmma_f16(v16h a, v16h b, v8f c) {
  return __builtin_amdgcn_wmma_f32_16x16x32_f16(
      false, a, false, b, (short)0, c, false, false);
}

// A fragment: 16x32 (MxK) f16, LDS row-major, leading dim `ld` (halves).
// lanes 0-15 = M rows; halves 0..15 = K g*16+0..15 (contiguous).
__device__ inline v16h load_a_frag(const _Float16* smem, int ld, int row0, int k0) {
  const int lane = threadIdx.x & 31;
  const int m = lane & 15, g = lane >> 4;
  const _Float16* base = smem + (row0 + m) * ld + k0 + g * 16;
  union { v16h v; v8h h[2]; } u;
  u.h[0] = *(const v8h*)(base);
  u.h[1] = *(const v8h*)(base + 8);
  return u.v;
}

// B fragment: 32x16 (KxN) f16 read from a TRANSPOSED LDS tile [n][k]:
// lane n = lane&15, lanes 16-31 hold K=16..31; contiguous 16 halves per lane.
__device__ inline v16h load_bt_frag(const _Float16* smem, int ld, int k0, int col0) {
  const int lane = threadIdx.x & 31;
  const int n = lane & 15, g = lane >> 4;
  const _Float16* base = smem + (col0 + n) * ld + k0 + g * 16;
  union { v16h v; v8h h[2]; } u;
  u.h[0] = *(const v8h*)(base);
  u.h[1] = *(const v8h*)(base + 8);
  return u.v;
}

// C/D 16x16 f32: VGPR r, lane l -> row = r + 8*(l>>4), col = l&15.
__device__ inline void store_tile(float* smem, int ld, int row0, int col0,
                                  v8f d, bool add) {
  const int lane = threadIdx.x & 31;
  const int n = lane & 15, g = lane >> 4;
#pragma unroll
  for (int r = 0; r < 8; ++r) {
    float* p = smem + (row0 + r + 8 * g) * ld + col0 + n;
    if (add) *p += d[r]; else *p = d[r];
  }
}

// ---------------------------------------------------------------------------
__global__ __launch_bounds__(256) void f32_to_f16_kernel(
    const float* __restrict__ in, _Float16* __restrict__ out, int n) {
  int i = blockIdx.x * 256 + threadIdx.x;
  if (i < n) out[i] = (_Float16)in[i];
}

// in: [K][N] f32 -> out: [N][K] f16 (transposed for B-operand staging)
__global__ __launch_bounds__(256) void transpose_f32_to_f16_kernel(
    const float* __restrict__ in, _Float16* __restrict__ out, int K, int N) {
  int i = blockIdx.x * 256 + threadIdx.x;
  if (i < K * N) {
    int k = i / N, n = i - k * N;           // coalesced read along n
    out[n * K + k] = (_Float16)in[i];
  }
}

// ---------------------------------------------------------------------------
// GEMM staging: 128x32-half A tile and 128x32-half Bt tile, 4 async b128
// instructions per thread (straight-line, no divergent loop).
// ---------------------------------------------------------------------------
__device__ inline void stage_gemm(_Float16* dA, _Float16* dB,
                                  const _Float16* a, const _Float16* b,
                                  int t0, int n0, int k0, int tid) {
  const int r = tid >> 2, c8 = (tid & 3) * 8;
  cp_b128(&dA[r * 40 + c8],        &a[(t0 + r) * C_DIM + k0 + c8]);
  cp_b128(&dA[(r + 64) * 40 + c8], &a[(t0 + r + 64) * C_DIM + k0 + c8]);
  cp_b128(&dB[r * 40 + c8],        &b[(n0 + r) * C_DIM + k0 + c8]);
  cp_b128(&dB[(r + 64) * 40 + c8], &b[(n0 + r + 64) * C_DIM + k0 + c8]);
}

// ---------------------------------------------------------------------------
// QKV GEMM: qkv = x @ w_qkv + b_qkv. 128x128 block tile, 8 waves, each wave
// a 32x64 strip (8 WMMA / k-step). Double-buffered async k-pipeline.
// ---------------------------------------------------------------------------
__device__ inline void qkv_scatter(int t, int j, float v,
                                   _Float16* __restrict__ Q,
                                   _Float16* __restrict__ K,
                                   _Float16* __restrict__ Vt) {
  if (j < C_DIM) {
    int hh = j >> 6, d = j & 63;
    Q[(hh * T_SEQ + t) * HD + d] = (_Float16)(v * 0.125f);  // 1/sqrt(64)
  } else if (j < 2 * C_DIM) {
    int jj = j - C_DIM, hh = jj >> 6, d = jj & 63;
    K[(hh * T_SEQ + t) * HD + d] = (_Float16)v;             // natural (h,t,d)
  } else {
    int jj = j - 2 * C_DIM, hh = jj >> 6, d = jj & 63;
    Vt[(hh * HD + d) * T_SEQ + t] = (_Float16)v;            // transposed (h,d,t)
  }
}

__global__ __launch_bounds__(256) void qkv_gemm_kernel(
    const _Float16* __restrict__ xh, const _Float16* __restrict__ wT,
    const float* __restrict__ bqkv,
    _Float16* __restrict__ Qh, _Float16* __restrict__ Kh,
    _Float16* __restrict__ Vth) {
  __shared__ __attribute__((aligned(16))) _Float16 sA[2][128 * 40];   // [m][k]
  __shared__ __attribute__((aligned(16))) _Float16 sBt[2][128 * 40];  // [n][k]
  const int t0 = blockIdx.x * 128;
  const int n0 = blockIdx.y * 128;
  const int tid = threadIdx.x;
  const int wave = tid >> 5, wr = wave >> 1, wc = wave & 1;
  v8f acc[2][4] = {};

  stage_gemm(sA[0], sBt[0], xh, wT, t0, n0, 0, tid);      // prefetch tile 0
  int cur = 0;
  for (int k0 = 0; k0 < C_DIM; k0 += 32) {
    const bool more = (k0 + 32) < C_DIM;
    if (more) {
      stage_gemm(sA[cur ^ 1], sBt[cur ^ 1], xh, wT, t0, n0, k0 + 32, tid);
      wait_async<4>();          // previous tile landed (in-order per wave)
    } else {
      wait_async<0>();
    }
    __syncthreads();
    v16h a0 = load_a_frag(sA[cur], 40, wr * 32, 0);
    v16h a1 = load_a_frag(sA[cur], 40, wr * 32 + 16, 0);
#pragma unroll
    for (int j = 0; j < 4; ++j) {
      v16h b = load_bt_frag(sBt[cur], 40, 0, wc * 64 + j * 16);
      acc[0][j] = wmma_f16(a0, b, acc[0][j]);
      acc[1][j] = wmma_f16(a1, b, acc[1][j]);
    }
    __syncthreads();            // reads done before buffer is re-staged
    cur ^= 1;
  }
  const int lane = tid & 31, n = lane & 15, g = lane >> 4;
#pragma unroll
  for (int i = 0; i < 2; ++i)
#pragma unroll
    for (int j = 0; j < 4; ++j)
#pragma unroll
      for (int r = 0; r < 8; ++r) {
        int t  = t0 + wr * 32 + i * 16 + r + 8 * g;
        int jj = n0 + wc * 64 + j * 16 + n;
        qkv_scatter(t, jj, acc[i][j][r] + bqkv[jj], Qh, Kh, Vth);
      }
}

// ---------------------------------------------------------------------------
// Landmark flash attention. Grid (32 query-sections, 12 heads), 256 threads.
// Bucket of key k is its section k>>6; landmarks + current section compete in
// bucket 63. prob(regular key in section s) = softmax_s(att) * p_landmark[s].
// Streamed: online softmax over {landmark scores, last-section scores};
// K/V section tiles double-buffered with async-to-LDS prefetch.
// ---------------------------------------------------------------------------
__device__ inline void stage_attn(_Float16* dK, _Float16* dVt,
                                  const _Float16* __restrict__ Kh,
                                  const _Float16* __restrict__ Vth,
                                  int hh, int kb, int tid) {
  const int r = tid >> 3, c8 = (tid & 7) * 8;
  cp_b128(&dK[r * 72 + c8],         &Kh[(hh * T_SEQ + kb + r) * HD + c8]);
  cp_b128(&dK[(r + 32) * 72 + c8],  &Kh[(hh * T_SEQ + kb + r + 32) * HD + c8]);
  cp_b128(&dVt[r * 72 + c8],        &Vth[(hh * HD + r) * T_SEQ + kb + c8]);
  cp_b128(&dVt[(r + 32) * 72 + c8], &Vth[(hh * HD + r + 32) * T_SEQ + kb + c8]);
}

__global__ __launch_bounds__(256) void landmark_attn_kernel(
    const _Float16* __restrict__ Qh, const _Float16* __restrict__ Kh,
    const _Float16* __restrict__ Vth, _Float16* __restrict__ Yh) {
  const int sq = blockIdx.x;
  const int hh = blockIdx.y;
  const int q0 = sq * 64;
  const int tid = threadIdx.x;
  const int wave = tid >> 5, wr = wave >> 1, wc = wave & 1;

  __shared__ __attribute__((aligned(16))) _Float16 sQ[64 * 72];      // [query][dim]
  __shared__ __attribute__((aligned(16))) _Float16 sK[2][64 * 72];   // [key][dim]
  __shared__ __attribute__((aligned(16))) _Float16 sVt[2][64 * 72];  // [dim][key]
  __shared__ __attribute__((aligned(16))) _Float16 sP[64 * 72];      // [query][key]
  __shared__ __attribute__((aligned(16))) float sS[64 * 68];
  __shared__ __attribute__((aligned(16))) float sAcc[64 * 68];
  __shared__ float m63[64], l63[64], alphaS[64], coefS[64], rowM[64];
  __shared__ float pmax[64 * 4], psum[64 * 4];

  {                                                 // Q tile (2 async/thread)
    const int r = tid >> 3, c8 = (tid & 7) * 8;
    cp_b128(&sQ[r * 72 + c8],        &Qh[(hh * T_SEQ + q0 + r) * HD + c8]);
    cp_b128(&sQ[(r + 32) * 72 + c8], &Qh[(hh * T_SEQ + q0 + r + 32) * HD + c8]);
  }
  stage_attn(sK[0], sVt[0], Kh, Vth, hh, 0, tid);   // prefetch section 0
  for (int i = tid; i < 64 * 68; i += 256) sAcc[i] = 0.0f;
  if (tid < 64) { m63[tid] = -__builtin_inff(); l63[tid] = 0.0f; }
  __syncthreads();

  int cur = 0;
  for (int s = 0; s <= sq; ++s) {
    if (s < sq) {
      stage_attn(sK[cur ^ 1], sVt[cur ^ 1], Kh, Vth, hh, (s + 1) * 64, tid);
      wait_async<4>();          // Q + section s landed; s+1 still in flight
    } else {
      wait_async<0>();
    }
    __syncthreads();
    const _Float16* cK  = sK[cur];
    const _Float16* cVt = sVt[cur];

    // ---- S = Q * K^T (64x64); per wave one 16x32 strip ------------------
    {
      v8f c0{}, c1{};
#pragma unroll
      for (int k0 = 0; k0 < HD; k0 += 32) {
        v16h a  = load_a_frag(sQ, 72, wr * 16, k0);
        v16h b0 = load_bt_frag(cK, 72, k0, wc * 32);
        v16h b1 = load_bt_frag(cK, 72, k0, wc * 32 + 16);
        c0 = wmma_f16(a, b0, c0);
        c1 = wmma_f16(a, b1, c1);
      }
      store_tile(sS, 68, wr * 16, wc * 32, c0, false);
      store_tile(sS, 68, wr * 16, wc * 32 + 16, c1, false);
    }
    __syncthreads();

    // ---- grouped softmax, streaming (4 threads per query row) -----------
    const bool lastsec = (s == sq);
    const int row = tid >> 2, qp = tid & 3, cb = qp * 16;
    {
      float lm = -__builtin_inff();
#pragma unroll
      for (int v4 = 0; v4 < 4; ++v4) {
        float4 sv = *(const float4*)&sS[row * 68 + cb + v4 * 4];
        const float* sp = (const float*)&sv;
#pragma unroll
        for (int e = 0; e < 4; ++e) {
          int c = cb + v4 * 4 + e;
          bool valid = lastsec ? (c <= row && c < 63) : (c < 63);
          if (valid) lm = fmaxf(lm, sp[e]);
        }
      }
      pmax[row * 4 + qp] = lm;
    }
    __syncthreads();
    if (qp == 0) {
      rowM[row] = fmaxf(fmaxf(pmax[row * 4], pmax[row * 4 + 1]),
                        fmaxf(pmax[row * 4 + 2], pmax[row * 4 + 3]));
    }
    __syncthreads();
    {
      float rm = rowM[row];
      float ls = 0.0f;
#pragma unroll
      for (int v4 = 0; v4 < 4; ++v4) {
        float4 sv = *(const float4*)&sS[row * 68 + cb + v4 * 4];
        const float* sp = (const float*)&sv;
#pragma unroll
        for (int e = 0; e < 4; ++e) {
          int c = cb + v4 * 4 + e;
          bool valid = lastsec ? (c <= row && c < 63) : (c < 63);
          if (valid) ls += __expf(sp[e] - rm);
        }
      }
      psum[row * 4 + qp] = ls;
    }
    __syncthreads();
    if (qp == 0) {
      float rsum = psum[row * 4] + psum[row * 4 + 1] +
                   psum[row * 4 + 2] + psum[row * 4 + 3];
      float rm = rowM[row];
      float mo = m63[row];
      if (!lastsec) {
        float z  = sS[row * 68 + 63];          // landmark score, section s
        float mn = fmaxf(mo, z);
        float al = __expf(mo - mn);
        float w  = __expf(z - mn);
        l63[row] = l63[row] * al + w;
        m63[row] = mn;
        alphaS[row] = al;
        coefS[row] = w / rsum;                 // gate * in-section normalize
      } else {
        float mn = fmaxf(mo, rm);
        float al = __expf(mo - mn);
        float cf = __expf(rm - mn);
        l63[row] = l63[row] * al + rsum * cf;
        m63[row] = mn;
        alphaS[row] = al;
        coefS[row] = cf;                       // direct bucket-63 weights
      }
    }
    __syncthreads();
#pragma unroll
    for (int ii = 0; ii < 4; ++ii) {            // 64x64 by 4-col vectors
      int i = ii * 256 + tid;
      int r = i >> 4, c4 = (i & 15) << 2;
      float cf = coefS[r], rm = rowM[r], al = alphaS[r];
      float4 sv = *(const float4*)&sS[r * 68 + c4];
      const float* sp = (const float*)&sv;
      v4h pk{};
#pragma unroll
      for (int e = 0; e < 4; ++e) {
        int c = c4 + e;
        bool valid = lastsec ? (c <= r && c < 63) : (c < 63);
        pk[e] = (_Float16)(valid ? cf * __expf(sp[e] - rm) : 0.0f);
      }
      *(v4h*)&sP[r * 72 + c4] = pk;
      float4* ap = (float4*)&sAcc[r * 68 + c4];
      float4 av = *ap;
      av.x *= al; av.y *= al; av.z *= al; av.w *= al;
      *ap = av;
    }
    __syncthreads();

    // ---- Acc += P * V ---------------------------------------------------
    {
      v8f c0{}, c1{};
#pragma unroll
      for (int k0 = 0; k0 < 64; k0 += 32) {
        v16h a  = load_a_frag(sP, 72, wr * 16, k0);
        v16h b0 = load_bt_frag(cVt, 72, k0, wc * 32);
        v16h b1 = load_bt_frag(cVt, 72, k0, wc * 32 + 16);
        c0 = wmma_f16(a, b0, c0);
        c1 = wmma_f16(a, b1, c1);
      }
      store_tile(sAcc, 68, wr * 16, wc * 32, c0, true);
      store_tile(sAcc, 68, wr * 16, wc * 32 + 16, c1, true);
    }
    __syncthreads();            // reads done before buffer re-staged
    cur ^= 1;
  }

#pragma unroll
  for (int ii = 0; ii < 4; ++ii) {
    int i = ii * 256 + tid;
    int r = i >> 4, c4 = (i & 15) << 2;
    float inv = 1.0f / l63[r];
    float4 av = *(const float4*)&sAcc[r * 68 + c4];
    v4h yk{};
    yk[0] = (_Float16)(av.x * inv); yk[1] = (_Float16)(av.y * inv);
    yk[2] = (_Float16)(av.z * inv); yk[3] = (_Float16)(av.w * inv);
    *(v4h*)&Yh[(q0 + r) * C_DIM + hh * HD + c4] = yk;
  }
}

// ---------------------------------------------------------------------------
// Output projection: out = Yh @ w_proj + b_proj (f32 out). 128x128 tiles,
// double-buffered async k-pipeline.
// ---------------------------------------------------------------------------
__global__ __launch_bounds__(256) void proj_gemm_kernel(
    const _Float16* __restrict__ yh, const _Float16* __restrict__ wT,
    const float* __restrict__ bproj, float* __restrict__ out) {
  __shared__ __attribute__((aligned(16))) _Float16 sA[2][128 * 40];
  __shared__ __attribute__((aligned(16))) _Float16 sBt[2][128 * 40];
  const int t0 = blockIdx.x * 128;
  const int n0 = blockIdx.y * 128;
  const int tid = threadIdx.x;
  const int wave = tid >> 5, wr = wave >> 1, wc = wave & 1;
  v8f acc[2][4] = {};

  stage_gemm(sA[0], sBt[0], yh, wT, t0, n0, 0, tid);
  int cur = 0;
  for (int k0 = 0; k0 < C_DIM; k0 += 32) {
    const bool more = (k0 + 32) < C_DIM;
    if (more) {
      stage_gemm(sA[cur ^ 1], sBt[cur ^ 1], yh, wT, t0, n0, k0 + 32, tid);
      wait_async<4>();
    } else {
      wait_async<0>();
    }
    __syncthreads();
    v16h a0 = load_a_frag(sA[cur], 40, wr * 32, 0);
    v16h a1 = load_a_frag(sA[cur], 40, wr * 32 + 16, 0);
#pragma unroll
    for (int j = 0; j < 4; ++j) {
      v16h b = load_bt_frag(sBt[cur], 40, 0, wc * 64 + j * 16);
      acc[0][j] = wmma_f16(a0, b, acc[0][j]);
      acc[1][j] = wmma_f16(a1, b, acc[1][j]);
    }
    __syncthreads();
    cur ^= 1;
  }
  const int lane = tid & 31, n = lane & 15, g = lane >> 4;
#pragma unroll
  for (int i = 0; i < 2; ++i)
#pragma unroll
    for (int j = 0; j < 4; ++j)
#pragma unroll
      for (int r = 0; r < 8; ++r) {
        int t  = t0 + wr * 32 + i * 16 + r + 8 * g;
        int jj = n0 + wc * 64 + j * 16 + n;
        out[t * C_DIM + jj] = acc[i][j][r] + bproj[jj];
      }
}

// ---------------------------------------------------------------------------
extern "C" void kernel_launch(void* const* d_in, const int* in_sizes, int n_in,
                              void* d_out, int out_size, void* d_ws, size_t ws_size,
                              hipStream_t stream) {
  (void)in_sizes; (void)n_in; (void)out_size; (void)ws_size;
  const float* x      = (const float*)d_in[0];
  // d_in[1] = is_mem: landmark pattern (t % 64 == 63) derived analytically.
  const float* w_qkv  = (const float*)d_in[2];
  const float* b_qkv  = (const float*)d_in[3];
  const float* w_proj = (const float*)d_in[4];
  const float* b_proj = (const float*)d_in[5];
  float* out = (float*)d_out;

  char* ws = (char*)d_ws;
  const size_t nX    = (size_t)T_SEQ * C_DIM;        // 1,572,864
  const size_t nWqkv = (size_t)C_DIM * 3 * C_DIM;    // 1,769,472
  const size_t nWp   = (size_t)C_DIM * C_DIM;        //   589,824
  const size_t nHead = (size_t)H_NUM * T_SEQ * HD;   // 1,572,864

  _Float16* xh     = (_Float16*)ws;  ws += nX * 2;
  _Float16* wqkvT  = (_Float16*)ws;  ws += nWqkv * 2;   // [2304][768]
  _Float16* wprojT = (_Float16*)ws;  ws += nWp * 2;     // [768][768]
  _Float16* Qh     = (_Float16*)ws;  ws += nHead * 2;   // (h,t,d) * 0.125
  _Float16* Kh     = (_Float16*)ws;  ws += nHead * 2;   // (h,t,d)
  _Float16* Vth    = (_Float16*)ws;  ws += nHead * 2;   // (h,d,t)
  _Float16* Yh     = (_Float16*)ws;  ws += nX * 2;      // (t, c)

  f32_to_f16_kernel<<<(int)((nX + 255) / 256), 256, 0, stream>>>(x, xh, (int)nX);
  transpose_f32_to_f16_kernel<<<(int)((nWqkv + 255) / 256), 256, 0, stream>>>(
      w_qkv, wqkvT, C_DIM, 3 * C_DIM);
  transpose_f32_to_f16_kernel<<<(int)((nWp + 255) / 256), 256, 0, stream>>>(
      w_proj, wprojT, C_DIM, C_DIM);

  qkv_gemm_kernel<<<dim3(T_SEQ / 128, 3 * C_DIM / 128), 256, 0, stream>>>(
      xh, wqkvT, b_qkv, Qh, Kh, Vth);

  landmark_attn_kernel<<<dim3(NSEC, H_NUM), 256, 0, stream>>>(Qh, Kh, Vth, Yh);

  proj_gemm_kernel<<<dim3(T_SEQ / 128, C_DIM / 128), 256, 0, stream>>>(
      Yh, wprojT, b_proj, out);
}
